// ScaleDotProductAttention_5557687681859
// MI455X (gfx1250) — matrix operand order repaired and checked by reading, not compiled, not run on hardware
//
#include <hip/hip_runtime.h>

// ---------------------------------------------------------------------------
// Ragged-masked scaled dot-product attention for MI455X (gfx1250, wave32).
// S = qk^T/8 (bf16 WMMA, fp32 acc) -> faithful mask (-1e-12) -> fp32 softmax
// -> score streamed out (nontemporal) -> out = P*V (bf16 WMMA, split-K, fp32).
// All WMMA executes under scalar (wave-uniform) control flow: EXEC all-1s.
// ---------------------------------------------------------------------------

#define LSEQ   900
#define DDIM   64
#define QTILE  16
#define NCTILES 57      // ceil(900/16)
#define SCOLS  928      // 29 * 32 : columns covered by the PV K-loop
#define SROW   932      // LDS row stride (floats); even (float2 ok), !%8 (bank spread)
#define PVKT   29       // K tiles of 32 covering 900 (pad cols -> prob exactly 0)

typedef __attribute__((ext_vector_type(16))) __bf16 v16bf;
typedef __attribute__((ext_vector_type(8)))  float  v8f;

// 16-bit A-matrix 16x32 lane layout (ISA 7.12.2): element e of the v16bf for
// a lane in half `hi` maps to K index:
__device__ __forceinline__ int a16_kidx(int e, int hi) {
    int vg = e >> 1, r = e & 1;
    int k = (vg < 4) ? (vg * 2 + r) : (16 + (vg - 4) * 2 + r);
    return k + (hi ? 8 : 0);
}

__global__ __launch_bounds__(256, 2)
void attn_ragged_kernel(const float* __restrict__ q,
                        const float* __restrict__ kmat,
                        const float* __restrict__ v,
                        const int*   __restrict__ dur,
                        float* __restrict__ outp,
                        float* __restrict__ score)
{
    __shared__ float Sbuf[QTILE * SROW];   // 59,648 B : 16 x 928 score strip (+pad)
    __shared__ float qs[QTILE * DDIM];     //  4,096 B : q tile, later PV reduce buf
    __shared__ float red[256];             //  1,024 B : reduction scratch
    __shared__ float rowmax_s[QTILE];
    __shared__ float rinv_s[QTILE];

    const int tid  = threadIdx.x;
    const int lane = tid & 31;
    // Compiler-visible wave-uniform wave id -> scalar branches; WMMA never
    // executes under a partial/zero EXEC mask (ISA 7.12 requirement).
    const int wave = __builtin_amdgcn_readfirstlane(tid >> 5);
    const int hi   = lane >> 4;   // which 16-lane half
    const int l16  = lane & 15;

    const int qt   = blockIdx.x;
    const int b    = blockIdx.y;
    const int q0   = qt * QTILE;
    const int durb = dur[b];

    // ---- stage q tile to LDS (row-clamped: rows >=900 give garbage that the
    //      mask step discards); init pad columns 912..927 to -inf ----
    {
        int row = tid >> 4;
        int col = (tid & 15) * 4;
        int r   = min(q0 + row, LSEQ - 1);
        float4 x = *(const float4*)(q + ((size_t)b * LSEQ + r) * DDIM + col);
        *(float4*)(qs + row * DDIM + col) = x;
        Sbuf[row * SROW + 912 + (tid & 15)] = -__builtin_inff();
    }
    __syncthreads();

    // ---- build bf16 A fragments of the q tile (K = 0..31, 32..63) ----
    v16bf aq[2];
#pragma unroll
    for (int kt = 0; kt < 2; ++kt) {
#pragma unroll
        for (int e = 0; e < 16; e += 2) {
            int k = kt * 32 + a16_kidx(e, hi);           // even -> float2 aligned
            float2 f = *(const float2*)(qs + l16 * DDIM + k);
            aq[kt][e]     = (__bf16)f.x;
            aq[kt][e + 1] = (__bf16)f.y;
        }
    }

    // ---- S = q k^T * 0.125, masked, into LDS strip (branch-free loads) ----
    for (int j = wave; j < NCTILES; j += 8) {
        const int c0 = j * QTILE;
        const int c  = c0 + l16;                   // this lane's column (N index)
        const int cl = min(c, LSEQ - 1);           // clamped: c>=900 -> -inf later
        v8f acc = {};
#pragma unroll
        for (int kt = 0; kt < 2; ++kt) {
            // B 32x16 layout: same K order as A (a16_kidx). B[kk][n] = K[b, c, kt*32 + a16_kidx(e, hi)].
            v16bf bk;
            const float* kp = kmat + ((size_t)b * LSEQ + cl) * DDIM
                                   + kt * 32 + hi * 8;
#pragma unroll
            for (int e4 = 0; e4 < 4; ++e4) {
                float4 f = *(const float4*)(kp + (e4 & 1) * 4 + (e4 >> 1) * 16);
                bk[e4 * 4 + 0] = (__bf16)f.x;
                bk[e4 * 4 + 1] = (__bf16)f.y;
                bk[e4 * 4 + 2] = (__bf16)f.z;
                bk[e4 * 4 + 3] = (__bf16)f.w;
            }
            acc = __builtin_amdgcn_wmma_f32_16x16x32_bf16(
                      false, aq[kt], false, bk, (short)0, acc, false, false);
        }
        // C/D layout: elem i -> row m = i + hi*8, col n = l16
#pragma unroll
        for (int i = 0; i < 8; ++i) {
            int m = i + hi * 8;
            int r = q0 + m;
            float s;
            if (c >= LSEQ)                   s = -__builtin_inff();  // pad col -> P=0
            else if (r < durb && c < durb)   s = acc[i] * 0.125f;    // 1/sqrt(64)
            else                             s = -1e-12f;            // faithful mask
            Sbuf[m * SROW + c] = s;
        }
    }
    __syncthreads();

    // ---- fp32 softmax over 928 cols (pads are -inf -> exp 0) ----
    const int m = tid >> 4;           // row 0..15
    const int p = tid & 15;           // 16 partials per row
    {
        float mx = -__builtin_inff();
        for (int c = p; c < SCOLS; c += 16)
            mx = fmaxf(mx, Sbuf[m * SROW + c]);
        red[m * 16 + p] = mx;
    }
    __syncthreads();
    if (p == 0) {
        float mx = red[m * 16];
#pragma unroll
        for (int j = 1; j < 16; ++j) mx = fmaxf(mx, red[m * 16 + j]);
        rowmax_s[m] = mx;
    }
    __syncthreads();
    {
        const float rm = rowmax_s[m];
        float sum = 0.f;
        for (int c = p; c < SCOLS; c += 16) {
            float e = __expf(Sbuf[m * SROW + c] - rm);
            Sbuf[m * SROW + c] = e;
            sum += e;
        }
        red[m * 16 + p] = sum;
    }
    __syncthreads();
    if (p == 0) {
        float s = 0.f;
#pragma unroll
        for (int j = 0; j < 16; ++j) s += red[m * 16 + j];
        rinv_s[m] = 1.0f / s;
    }
    __syncthreads();
    {
        const float ri = rinv_s[m];
        const int   r  = q0 + m;
        float* srow = score + ((size_t)b * LSEQ + r) * LSEQ;
        for (int c = p; c < SCOLS; c += 16) {
            float pv = Sbuf[m * SROW + c] * ri;
            Sbuf[m * SROW + c] = pv;                    // probabilities for PV
            if (c < LSEQ && r < LSEQ)
                __builtin_nontemporal_store(pv, srow + c);  // 207MB write-once stream
        }
    }
    __syncthreads();

    // ---- out = P * V : 8 waves, split-K. wave = {khalf[1], ntile[1:0]} ----
    const int nt  = wave & 3;           // N tile (16 cols of D=64)
    const int kh  = wave >> 2;          // K half
    const int n0  = nt * 16;
    const int nn  = n0 + l16;
    const int kt0 = kh ? 15 : 0;
    const int kt1 = kh ? PVKT : 15;
    v8f acc = {};
    for (int ktile = kt0; ktile < kt1; ++ktile) {
        const int kbase = ktile * 32;
        // A fragment from LDS probabilities (row m = l16); cols >=900 are 0.
        v16bf ap;
#pragma unroll
        for (int e = 0; e < 16; e += 2) {
            int k = kbase + a16_kidx(e, hi);
            float2 f = *(const float2*)(Sbuf + l16 * SROW + k);
            ap[e]     = (__bf16)f.x;
            ap[e + 1] = (__bf16)f.y;
        }
        // B fragment: B[e][n] = V[b, kbase + a16_kidx(e, hi), nn]; rows k>=900 are
        // multiplied by P==0, so clamp the address instead of branching.
        v16bf bv;
#pragma unroll
        for (int e = 0; e < 16; ++e) {
            int k = min(kbase + a16_kidx(e, hi), LSEQ - 1);
            bv[e] = (__bf16)v[((size_t)b * LSEQ + k) * DDIM + nn];
        }
        acc = __builtin_amdgcn_wmma_f32_16x16x32_bf16(
                  false, ap, false, bv, (short)0, acc, false, false);
    }
    __syncthreads();                    // Sbuf/qs reads done; qs reusable
    if (kh == 1) {                      // upper K half parks partials in qs
#pragma unroll
        for (int i = 0; i < 8; ++i)
            qs[(i + hi * 8) * DDIM + nn] = acc[i];
    }
    __syncthreads();
    if (kh == 0) {                      // lower K half reduces + writes out
#pragma unroll
        for (int i = 0; i < 8; ++i) {
            int mm = i + hi * 8;
            int r  = q0 + mm;
            float val = acc[i] + qs[mm * DDIM + nn];
            if (r < LSEQ)
                outp[((size_t)b * LSEQ + r) * DDIM + n0 + l16] = val;
        }
    }
}

extern "C" void kernel_launch(void* const* d_in, const int* in_sizes, int n_in,
                              void* d_out, int out_size, void* d_ws, size_t ws_size,
                              hipStream_t stream) {
    const float* q    = (const float*)d_in[0];
    const float* kmat = (const float*)d_in[1];
    const float* v    = (const float*)d_in[2];
    const int*   dur  = (const int*)d_in[3];

    const int B = in_sizes[3];                       // 64
    float* outp  = (float*)d_out;                    // [B, L, D]
    float* score = outp + (size_t)B * LSEQ * DDIM;   // [B, L, L]

    dim3 grid(NCTILES, B);
    dim3 block(256);
    attn_ragged_kernel<<<grid, block, 0, stream>>>(q, kmat, v, dur, outp, score);
}
